// FasterMultiHeadAttention_43533788512396
// MI455X (gfx1250) — compile-verified, matrix operand order checked by reading, not run on hardware
//
#include <hip/hip_runtime.h>

// ---------------------------------------------------------------------------
// MHA forward for MI455X (gfx1250, wave32, WMMA), register-blocked GEMMs +
// block-cooperative flash attention with async global->LDS K/V staging.
// Shapes: B=2, S=2048, D=1024, H=16, dh=64.
// ---------------------------------------------------------------------------

typedef __bf16 bf16;
typedef __attribute__((ext_vector_type(16))) __bf16 v16bf;
typedef __attribute__((ext_vector_type(8)))  __bf16 v8bf;
typedef __attribute__((ext_vector_type(8)))  float  v8f;
typedef int v4i_g __attribute__((vector_size(16)));   // int32x4 (gcc-style)

#define BATCH   2
#define SEQ     2048
#define DMODEL  1024
#define NHEADS  16
#define DHEAD   64
#define NH      (BATCH * NHEADS)   // 32 head-batches
#define ROWS    (BATCH * SEQ)      // 4096
#define D3      (3 * DMODEL)       // 3072

// CDNA5 async global->LDS copy path (ASYNCcnt), when the toolchain has it.
#if defined(__HIP_DEVICE_COMPILE__) &&                                   \
    __has_builtin(__builtin_amdgcn_global_load_async_to_lds_b128) &&     \
    __has_builtin(__builtin_amdgcn_s_wait_asynccnt)
#define HAVE_ASYNC_LDS 1
#else
#define HAVE_ASYNC_LDS 0
#endif

static __device__ inline v8f zero8() {
  v8f z;
#pragma unroll
  for (int i = 0; i < 8; ++i) z[i] = 0.0f;
  return z;
}

// D = A*B + C, bf16 inputs, f32 accumulate (v_wmma_f32_16x16x32_bf16)
static __device__ inline v8f wmma_bf16(v16bf a, v16bf b, v8f c) {
  return __builtin_amdgcn_wmma_f32_16x16x32_bf16(
      /*neg_a=*/false, a, /*neg_b=*/false, b,
      /*c_mod=*/(short)0, c, /*reuse_a=*/false, /*reuse_b=*/false);
}

// A (or B with memory layout [n][k]) fragment, 16x32 bf16, row-major source.
// ISA layout: lanes 0-15 -> M=lane, K = kh+{0..7} in v0..3, K = 16+kh+{0..7}
// in v4..7, with kh = 0 (lanes 0-15) or 8 (lanes 16-31).
static __device__ inline v16bf load_frag_rm(const bf16* __restrict__ A, long lda,
                                            int row0, int k0) {
  int l  = (int)(threadIdx.x & 31u);
  int m  = l & 15;
  int kh = (l >> 4) << 3;
  const bf16* p = A + (long)(row0 + m) * lda + k0 + kh;
  v8bf lo = *(const v8bf*)(p);        // K = kh .. kh+7     (16B aligned)
  v8bf hi = *(const v8bf*)(p + 16);   // K = kh+16 .. kh+23
  v16bf r;
#pragma unroll
  for (int i = 0; i < 8; ++i) { r[i] = lo[i]; r[i + 8] = hi[i]; }
  return r;
}

// A fragment from an fp32 row-major source, converting to bf16 on the fly.
static __device__ inline v16bf load_frag_f32(const float* __restrict__ A, long lda,
                                             int row0, int k0) {
  int l  = (int)(threadIdx.x & 31u);
  int m  = l & 15;
  int kh = (l >> 4) << 3;
  const float* p = A + (long)(row0 + m) * lda + k0 + kh;
  v16bf r;
#pragma unroll
  for (int i = 0; i < 8; ++i) { r[i] = (bf16)p[i]; r[i + 8] = (bf16)p[i + 16]; }
  return r;
}

// B fragment (32x16) from a K-major source: element(k,n) = M[(k0+k)*ld + n0+n].
static __device__ inline v16bf load_frag_km(const bf16* __restrict__ M, long ld,
                                            int k0, int n0) {
  int l  = (int)(threadIdx.x & 31u);
  int n  = l & 15;
  int kh = (l >> 4) << 3;
  v16bf r;
#pragma unroll
  for (int i = 0; i < 8; ++i) {
    r[i]     = M[(long)(k0 + kh + i) * ld + n0 + n];
    r[i + 8] = M[(long)(k0 + kh + 16 + i) * ld + n0 + n];
  }
  return r;
}

// One 16-byte chunk global->LDS, async when available.
static __device__ inline void copy16_g2l(bf16* ldst, const bf16* gsrc) {
#if HAVE_ASYNC_LDS
  __builtin_amdgcn_global_load_async_to_lds_b128(
      (__attribute__((address_space(1))) v4i_g*)gsrc,
      (__attribute__((address_space(3))) v4i_g*)ldst,
      /*imm offset=*/0, /*cpol=*/0);
#else
  *(v8bf*)ldst = *(const v8bf*)gsrc;
#endif
}

static __device__ inline void wait_async_copies() {
#if HAVE_ASYNC_LDS
  __builtin_amdgcn_s_wait_asynccnt(0);
#endif
}

// ---------------------------------------------------------------------------
// Kernel 1: in-proj GEMM  proj[4096,3072] = X_bf16 @ W_in^T + b_in.
// One wave owns a 64x64 C tile: 16 v8f accumulators, 16 WMMAs per k-step,
// each fragment load feeds 4 WMMAs. Epilogue scatters the interleaved
// (head, d_head, qkv) channels into Q/K/V workspaces [B*H, S, dh] bf16.
// ---------------------------------------------------------------------------
__global__ __launch_bounds__(256) void qkv_proj_kernel(
    const float* __restrict__ x, const bf16* __restrict__ Win,
    const bf16* __restrict__ bin,
    bf16* __restrict__ Q, bf16* __restrict__ K, bf16* __restrict__ V) {
  int w    = blockIdx.x * 8 + (threadIdx.x >> 5);
  int mt   = w / (D3 / 64);                // 64 row-tiles
  int nt   = w % (D3 / 64);                // 48 col-tiles
  int row0 = mt * 64, col0 = nt * 64;

  v8f acc[4][4];
#pragma unroll
  for (int mi = 0; mi < 4; ++mi)
#pragma unroll
    for (int ni = 0; ni < 4; ++ni) acc[mi][ni] = zero8();

  for (int k0 = 0; k0 < DMODEL; k0 += 32) {
    v16bf a[4];
#pragma unroll
    for (int mi = 0; mi < 4; ++mi)
      a[mi] = load_frag_f32(x, DMODEL, row0 + mi * 16, k0);
#pragma unroll
    for (int ni = 0; ni < 4; ++ni) {
      v16bf b = load_frag_rm(Win, DMODEL, col0 + ni * 16, k0);
#pragma unroll
      for (int mi = 0; mi < 4; ++mi)
        acc[mi][ni] = wmma_bf16(a[mi], b, acc[mi][ni]);
    }
  }

  int l  = (int)(threadIdx.x & 31u);
  int nc = l & 15;
  int mh = (l >> 4) << 3;
#pragma unroll
  for (int ni = 0; ni < 4; ++ni) {
    int o = col0 + ni * 16 + nc;            // output channel
    float bias = (float)bin[o];
    int c  = o % 3;                         // 0=Q 1=K 2=V
    int dd = (o / 3) % DHEAD;
    int h  = o / (3 * DHEAD);
    bf16* dst = (c == 0) ? Q : (c == 1) ? K : V;
#pragma unroll
    for (int mi = 0; mi < 4; ++mi) {
#pragma unroll
      for (int j = 0; j < 8; ++j) {
        int r  = row0 + mi * 16 + mh + j;   // flat (b,s) row
        int bi = r >> 11;                   // r / SEQ
        int s  = r & (SEQ - 1);
        long idx = (((long)(bi * NHEADS + h)) * SEQ + s) * DHEAD + dd;
        dst[idx] = (bf16)(acc[mi][ni][j] + bias);
      }
    }
  }
}

// ---------------------------------------------------------------------------
// Kernel 2: causal flash attention, block-cooperative.
// A block's 8 waves all work on the SAME head-batch, on 8 consecutive
// 32-row q tiles (256 q rows). Each 32-key K/V tile (4 KB each) is staged
// once per block into LDS with async global->LDS copies and shared by all
// waves. Per-wave compute is causally guarded (wave-uniform branch).
// ---------------------------------------------------------------------------
__global__ __launch_bounds__(256) void attn_kernel(
    const bf16* __restrict__ Q, const bf16* __restrict__ K,
    const bf16* __restrict__ V, const float* __restrict__ mask,
    bf16* __restrict__ merged) {
  __shared__ bf16 ldsK[32][DHEAD];          // 4 KB, shared K tile
  __shared__ bf16 ldsV[32][DHEAD];          // 4 KB, shared V tile
  __shared__ bf16 ldsP[8][2][16 * 32];      // per-wave P staging, 16 KB

  int wid  = threadIdx.x >> 5;
  int n    = blockIdx.x >> 3;               // 0..31 head-batch
  int qblk = blockIdx.x & 7;                // 8 q-blocks of 256 rows
  int q0   = qblk * 256 + wid * 32;         // this wave's 32 q rows

  const bf16* Qh = Q + (long)n * SEQ * DHEAD;
  const bf16* Kh = K + (long)n * SEQ * DHEAD;
  const bf16* Vh = V + (long)n * SEQ * DHEAD;

  int l  = (int)(threadIdx.x & 31u);
  int nc = l & 15;
  int mh = (l >> 4) << 3;
  int kh = (l >> 4) << 3;

  // staging assignment: thread t copies one 16B chunk of K and of V
  int srow = (int)(threadIdx.x >> 3);        // 0..31
  int scol = (int)((threadIdx.x & 7u) * 8);  // 0..56 step 8 elems (16 B)

  // Q fragments: 2 row-tiles x 2 k-halves of dh.
  v16bf aQ[2][2];
#pragma unroll
  for (int r = 0; r < 2; ++r)
#pragma unroll
    for (int kk = 0; kk < 2; ++kk)
      aQ[r][kk] = load_frag_rm(Qh, DHEAD, q0 + r * 16, kk * 32);

  float run_max[2][8], run_sum[2][8];
  v8f O[2][4];
#pragma unroll
  for (int r = 0; r < 2; ++r) {
#pragma unroll
    for (int j = 0; j < 8; ++j) { run_max[r][j] = -3.0e38f; run_sum[r][j] = 0.0f; }
#pragma unroll
    for (int t = 0; t < 4; ++t) O[r][t] = zero8();
  }

  const float scale = 0.125f;                // 1/sqrt(64)
  const int numKB = qblk * 8 + 8;            // block-wide causal bound
  const int myEnd = q0 + 32;                 // keys this wave needs (< myEnd)

  for (int kb = 0; kb < numKB; ++kb) {
    int kc = kb * 32;

    // ---- stage K/V 32-key tiles into LDS (whole block cooperates) ----
    copy16_g2l(&ldsK[srow][scol], Kh + (long)(kc + srow) * DHEAD + scol);
    copy16_g2l(&ldsV[srow][scol], Vh + (long)(kc + srow) * DHEAD + scol);
    wait_async_copies();
    __syncthreads();

    if (kc < myEnd) {                        // wave-uniform causal guard
      // ---- S = Q @ K^T: 2 row-tiles x 2 key-col tiles, k = dh = 64 ----
      v8f s[2][2];
#pragma unroll
      for (int r = 0; r < 2; ++r)
#pragma unroll
        for (int t = 0; t < 2; ++t) s[r][t] = zero8();

#pragma unroll
      for (int t = 0; t < 2; ++t) {
#pragma unroll
        for (int kk = 0; kk < 2; ++kk) {
          v16bf bK = load_frag_rm(&ldsK[0][0], DHEAD, t * 16, kk * 32);
#pragma unroll
          for (int r = 0; r < 2; ++r)
            s[r][t] = wmma_bf16(aQ[r][kk], bK, s[r][t]);
        }
      }

      // ---- scale + additive causal mask (fp32), online softmax ----
#pragma unroll
      for (int r = 0; r < 2; ++r) {
#pragma unroll
        for (int j = 0; j < 8; ++j) {
          long mi  = (long)(q0 + r * 16 + mh + j) * SEQ + kc + nc;
          float f0 = s[r][0][j] * scale + mask[mi];
          float f1 = s[r][1][j] * scale + mask[mi + 16];

          float bm = fmaxf(f0, f1);
          bm = fmaxf(bm, __shfl_xor(bm, 1, 32));
          bm = fmaxf(bm, __shfl_xor(bm, 2, 32));
          bm = fmaxf(bm, __shfl_xor(bm, 4, 32));
          bm = fmaxf(bm, __shfl_xor(bm, 8, 32));

          float nm   = fmaxf(run_max[r][j], bm);
          float corr = __expf(run_max[r][j] - nm);
          run_max[r][j] = nm;

          float p0 = __expf(f0 - nm);
          float p1 = __expf(f1 - nm);
          float bs = p0 + p1;
          bs += __shfl_xor(bs, 1, 32);
          bs += __shfl_xor(bs, 2, 32);
          bs += __shfl_xor(bs, 4, 32);
          bs += __shfl_xor(bs, 8, 32);
          run_sum[r][j] = run_sum[r][j] * corr + bs;

#pragma unroll
          for (int t = 0; t < 4; ++t) O[r][t][j] *= corr;

          ldsP[wid][r][(mh + j) * 32 + nc]      = (bf16)p0;
          ldsP[wid][r][(mh + j) * 32 + nc + 16] = (bf16)p1;
        }
      }

      // ---- reload P as A fragments (C-layout -> A-layout via LDS) ----
      v16bf aP[2];
#pragma unroll
      for (int r = 0; r < 2; ++r) {
        const bf16* p = &ldsP[wid][r][(l & 15) * 32];
#pragma unroll
        for (int i = 0; i < 8; ++i) {
          aP[r][i]     = p[kh + i];
          aP[r][i + 8] = p[kh + 16 + i];
        }
      }

      // ---- O += P (16x32) @ V (32x64): V fragments feed both row-tiles ----
#pragma unroll
      for (int vt = 0; vt < 4; ++vt) {
        v16bf bV = load_frag_km(&ldsV[0][0], DHEAD, 0, vt * 16);
#pragma unroll
        for (int r = 0; r < 2; ++r)
          O[r][vt] = wmma_bf16(aP[r], bV, O[r][vt]);
      }
    }

    __syncthreads();                         // protect ldsK/ldsV reuse
  }

  // ---- normalize and write merged[B, S, H*dh] ----
  int bi = n >> 4;
  int h  = n & 15;
#pragma unroll
  for (int r = 0; r < 2; ++r) {
#pragma unroll
    for (int j = 0; j < 8; ++j) {
      float inv = 1.0f / run_sum[r][j];
      int s = q0 + r * 16 + mh + j;
      long base = ((long)(bi * SEQ + s)) * DMODEL + h * DHEAD + nc;
#pragma unroll
      for (int vt = 0; vt < 4; ++vt)
        merged[base + vt * 16] = (bf16)(O[r][vt][j] * inv);
    }
  }
}

// ---------------------------------------------------------------------------
// Kernel 3: out-proj GEMM  out[4096,1024] = merged @ W_out^T + b_out.
// Same 64x64 per-wave register blocking as kernel 1.
// ---------------------------------------------------------------------------
__global__ __launch_bounds__(256) void out_proj_kernel(
    const bf16* __restrict__ merged, const bf16* __restrict__ Wout,
    const bf16* __restrict__ bout, bf16* __restrict__ out) {
  int w    = blockIdx.x * 8 + (threadIdx.x >> 5);
  int mt   = w / (DMODEL / 64);            // 64 row-tiles
  int nt   = w % (DMODEL / 64);            // 16 col-tiles
  int row0 = mt * 64, col0 = nt * 64;

  v8f acc[4][4];
#pragma unroll
  for (int mi = 0; mi < 4; ++mi)
#pragma unroll
    for (int ni = 0; ni < 4; ++ni) acc[mi][ni] = zero8();

  for (int k0 = 0; k0 < DMODEL; k0 += 32) {
    v16bf a[4];
#pragma unroll
    for (int mi = 0; mi < 4; ++mi)
      a[mi] = load_frag_rm(merged, DMODEL, row0 + mi * 16, k0);
#pragma unroll
    for (int ni = 0; ni < 4; ++ni) {
      v16bf b = load_frag_rm(Wout, DMODEL, col0 + ni * 16, k0);
#pragma unroll
      for (int mi = 0; mi < 4; ++mi)
        acc[mi][ni] = wmma_bf16(a[mi], b, acc[mi][ni]);
    }
  }

  int l  = (int)(threadIdx.x & 31u);
  int nc = l & 15;
  int mh = (l >> 4) << 3;
#pragma unroll
  for (int ni = 0; ni < 4; ++ni) {
    int o = col0 + ni * 16 + nc;
    float bias = (float)bout[o];
#pragma unroll
    for (int mi = 0; mi < 4; ++mi) {
#pragma unroll
      for (int j = 0; j < 8; ++j) {
        int r = row0 + mi * 16 + mh + j;
        out[(long)r * DMODEL + o] = (bf16)(acc[mi][ni][j] + bias);
      }
    }
  }
}

// ---------------------------------------------------------------------------
extern "C" void kernel_launch(void* const* d_in, const int* in_sizes, int n_in,
                              void* d_out, int out_size, void* d_ws, size_t ws_size,
                              hipStream_t stream) {
  (void)in_sizes; (void)n_in; (void)out_size; (void)ws_size;

  const float* x    = (const float*)d_in[0];
  const float* mask = (const float*)d_in[1];
  const bf16*  Win  = (const bf16*)d_in[2];
  const bf16*  bin  = (const bf16*)d_in[3];
  const bf16*  Wout = (const bf16*)d_in[4];
  const bf16*  bout = (const bf16*)d_in[5];
  bf16* out = (bf16*)d_out;

  char* ws = (char*)d_ws;
  const size_t per = (size_t)NH * SEQ * DHEAD * sizeof(bf16);  // 8 MB each
  bf16* Q      = (bf16*)(ws);
  bf16* K      = (bf16*)(ws + per);
  bf16* V      = (bf16*)(ws + 2 * per);
  bf16* merged = (bf16*)(ws + 3 * per);

  dim3 blk(256);
  // 1) QKV projection: 64 x 48 tiles of 64x64, 8 waves/block
  qkv_proj_kernel<<<(ROWS / 64) * (D3 / 64) / 8, blk, 0, stream>>>(
      x, Win, bin, Q, K, V);
  // 2) Attention: 32 head-batches x 8 q-blocks of 256 rows
  attn_kernel<<<NH * (SEQ / 256), blk, 0, stream>>>(
      Q, K, V, mask, merged);
  // 3) Output projection: 64 x 16 tiles of 64x64
  out_proj_kernel<<<(ROWS / 64) * (DMODEL / 64) / 8, blk, 0, stream>>>(
      merged, Wout, bout, out);
}
